// NTTailAttentionModel2Softmax_2611340116245
// MI455X (gfx1250) — matrix operand order repaired and checked by reading, not compiled, not run on hardware
//
#include <hip/hip_runtime.h>

static constexpr int kSEQ = 1024, kB = 64, kE = 256, kH = 512, kV = 10000;
static constexpr int kNWG = 32;          // persistent workgroups for the LSTM

typedef __attribute__((ext_vector_type(16))) __bf16 v16bf;
typedef __attribute__((ext_vector_type(8)))  __bf16 v8bf;
typedef __attribute__((ext_vector_type(8)))  float  v8f;

// ---------------- workspace layout (bytes) ----------------
static constexpr size_t CTR_OFF  = 0;                                    // 256 B (barrier counter)
static constexpr size_t XB_OFF   = 256;                                  // bf16 [SEQ][B][E]
static constexpr size_t OUTS_OFF = XB_OFF  + (size_t)kSEQ*kB*kE*2;       // bf16 [SEQ][B][H]
static constexpr size_t HB_OFF   = OUTS_OFF+ (size_t)kSEQ*kB*kH*2;       // bf16 [2][B][H] (double buffer)
static constexpr size_t AV_OFF   = HB_OFF  + (size_t)2*kB*kH*2;          // f32 [H]
static constexpr size_t SC_OFF   = AV_OFF  + 4096;                       // f32 [SEQ-1][B]
static constexpr size_t CAT_OFF  = SC_OFF  + 262144;                     // bf16 [B][2H]

// ---------------- WMMA fragment helpers ----------------
// A (16x32 bf16, row-major source): caller passes p = &src[m*ld + kbase + hi*8]
__device__ __forceinline__ v16bf load_a16(const __bf16* p) {
  v16bf r;
  v8bf lo = *(const v8bf*)(p);        // K = kbase + hi*8 + 0..7
  v8bf h8 = *(const v8bf*)(p + 16);   // K = kbase + hi*8 + 16..23
#pragma unroll
  for (int i = 0; i < 8; ++i) { r[i] = lo[i]; r[8 + i] = h8[i]; }
  return r;
}

// B fragment pre-swizzled in LDS: frag stride 512 elems, lane stride 16 elems
__device__ __forceinline__ v16bf lds_b(const __bf16* Bsw, int frag, int lane) {
  const __bf16* p = Bsw + (size_t)frag * 512 + (size_t)lane * 16;
  v8bf lo = *(const v8bf*)(p);
  v8bf h8 = *(const v8bf*)(p + 8);
  v16bf r;
#pragma unroll
  for (int i = 0; i < 8; ++i) { r[i] = lo[i]; r[8 + i] = h8[i]; }
  return r;
}

// B fragment from f32 row-major weights: p = &w[n*ld + kbase + hi*8], f32 at p[0..7] and p[16..23]
__device__ __forceinline__ v16bf load_b_f32(const float* p) {
  v16bf r;
#pragma unroll
  for (int i = 0; i < 16; i += 4) {
    float4 v = *(const float4*)(p + i + (i >= 8 ? 8 : 0));
    r[i] = (__bf16)v.x; r[i+1] = (__bf16)v.y; r[i+2] = (__bf16)v.z; r[i+3] = (__bf16)v.w;
  }
  return r;
}

__device__ __forceinline__ v8f wmma_bf16(v16bf a, v16bf b, v8f c) {
  return __builtin_amdgcn_wmma_f32_16x16x32_bf16(false, a, false, b, (short)0, c, false, false);
}

__device__ __forceinline__ float sigf(float x) { return 1.0f / (1.0f + __expf(-x)); }

// ---------------- kernels ----------------
__global__ void embed_bf16_k(const int* __restrict__ idx, const float* __restrict__ emb,
                             __bf16* __restrict__ Xb) {
  int gid = blockIdx.x * 256 + threadIdx.x;          // SEQ*B*(E/4) threads
  int e4  = (gid & 63) * 4;
  int sb  = gid >> 6;
  int tok = idx[sb];
  float4 v = *(const float4*)(emb + (size_t)tok * kE + e4);
  __bf16* o = Xb + (size_t)sb * kE + e4;
  o[0] = (__bf16)v.x; o[1] = (__bf16)v.y; o[2] = (__bf16)v.z; o[3] = (__bf16)v.w;
}

__global__ void init_h_k(const float* __restrict__ h0, __bf16* __restrict__ Hb) {
  int i = blockIdx.x * 256 + threadIdx.x;            // B*H threads
  Hb[i] = (__bf16)h0[i];
}

__global__ void lstm_persistent_k(const float* __restrict__ w_ih, const float* __restrict__ w_hh,
                                  const float* __restrict__ b_ih, const float* __restrict__ b_hh,
                                  const float* __restrict__ c0,
                                  const __bf16* __restrict__ Xb, __bf16* __restrict__ Hb,
                                  __bf16* __restrict__ outs,
                                  float* __restrict__ hT, float* __restrict__ cT,
                                  unsigned* __restrict__ ctr) {
  extern __shared__ char smem[];
  __bf16* Bsw = (__bf16*)smem;                 // 96 KB: 96 frags * 512 bf16 (24 ksteps * 4 ntiles)
  float*  Gs  = (float*)(smem + 96 * 1024);    // 16 KB: 64x64 gate tile

  const int wg   = blockIdx.x;                 // 0..31 -> h columns [wg*16, wg*16+16)
  const int t    = threadIdx.x;
  const int lane = t & 31, wave = t >> 5;
  const int hi   = lane >> 4, ml = lane & 15;

  // ---- stage weight slice into LDS, fragment-swizzled, f32 -> bf16 ----
  for (int fi = t; fi < 96 * 512; fi += 256) {
    int frag = fi >> 9, r = fi & 511;
    int l = r >> 4, e = r & 15;
    int kk = frag >> 2, ni = frag & 3;
    int k  = kk * 32 + (e & 7) + ((l >> 4) << 3) + ((e >> 3) << 4);    // K = (e&7) + 8*hi + 16*(e>>3) within 32-K tile
    int n  = ni * 512 + wg * 16 + (l & 15);    // gate group ni, column in slice
    float wv = (k < kE) ? w_ih[(size_t)n * kE + k] : w_hh[(size_t)n * kH + (k - kE)];
    Bsw[fi] = (__bf16)wv;
  }

  // ---- per-thread elementwise state: thread t owns (b = t/16 + 16q, jl = t&15) ----
  const int jl   = t & 15;
  const int hcol = wg * 16 + jl;
  float creg[4], biasv[4];
#pragma unroll
  for (int q = 0; q < 4; ++q) creg[q] = c0[(size_t)((t >> 4) + 16 * q) * kH + hcol];
#pragma unroll
  for (int g = 0; g < 4; ++g) { int n = g * kH + hcol; biasv[g] = b_ih[n] + b_hh[n]; }

  const int mi = wave >> 1, nb = (wave & 1) * 2;   // each wave: 1 M-tile x 2 N-tiles
  const size_t arowoff = (size_t)(mi * 16 + ml) * kE + hi * 8;
  const v8f vzero = {};
  __syncthreads();

  // ---- prologue: X-part of step 0 (independent of recurrence) ----
  v8f aX0 = vzero, aX1 = vzero;
  {
    const __bf16* arow = Xb + arowoff;
#pragma unroll
    for (int kk = 0; kk < 8; ++kk) {
      v16bf a = load_a16(arow + kk * 32);
      aX0 = wmma_bf16(a, lds_b(Bsw, kk * 4 + nb, lane), aX0);
      aX1 = wmma_bf16(a, lds_b(Bsw, kk * 4 + nb + 1, lane), aX1);
    }
  }

  for (int s = 0; s < kSEQ; ++s) {
    // ---- wait: all WGs finished step s-1, Hb[s&1] is valid ----
    if (s > 0) {
      if (t == 0) {
        unsigned target = (unsigned)(kNWG * s);
        while (__hip_atomic_load(ctr, __ATOMIC_RELAXED, __HIP_MEMORY_SCOPE_AGENT) < target)
          __builtin_amdgcn_s_sleep(1);
      }
      __syncthreads();
      __threadfence();
    }

    // ---- H-part: 16 k-steps split into two independent accumulation chains ----
    v8f aH0a = vzero, aH0b = vzero, aH1a = vzero, aH1b = vzero;
    const __bf16* hrow = Hb + (size_t)(s & 1) * (kB * kH) + (size_t)(mi * 16 + ml) * kH + hi * 8;
#pragma unroll
    for (int kk = 0; kk < 16; kk += 2) {
      v16bf a0 = load_a16(hrow + kk * 32);
      v16bf a1 = load_a16(hrow + (kk + 1) * 32);
      aH0a = wmma_bf16(a0, lds_b(Bsw, (8 + kk) * 4 + nb, lane), aH0a);
      aH1a = wmma_bf16(a0, lds_b(Bsw, (8 + kk) * 4 + nb + 1, lane), aH1a);
      aH0b = wmma_bf16(a1, lds_b(Bsw, (9 + kk) * 4 + nb, lane), aH0b);
      aH1b = wmma_bf16(a1, lds_b(Bsw, (9 + kk) * 4 + nb + 1, lane), aH1b);
    }
#pragma unroll
    for (int r = 0; r < 8; ++r) {                  // C layout: M = r + 8*hi, N = ml
      Gs[(mi * 16 + r + hi * 8) * 64 + nb * 16 + ml]       = aX0[r] + aH0a[r] + aH0b[r];
      Gs[(mi * 16 + r + hi * 8) * 64 + (nb + 1) * 16 + ml] = aX1[r] + aH1a[r] + aH1b[r];
    }
    __syncthreads();

    __bf16* Hn = Hb + (size_t)((s + 1) & 1) * (kB * kH);
#pragma unroll
    for (int q = 0; q < 4; ++q) {
      int b = (t >> 4) + 16 * q;
      float xi = Gs[b * 64 + jl]       + biasv[0];
      float xf = Gs[b * 64 + 16 + jl]  + biasv[1];
      float xg = Gs[b * 64 + 32 + jl]  + biasv[2];
      float xo = Gs[b * 64 + 48 + jl]  + biasv[3];
      float c  = sigf(xf) * creg[q] + sigf(xi) * tanhf(xg);
      float h  = sigf(xo) * tanhf(c);
      creg[q]  = c;
      __bf16 hb = (__bf16)h;
      Hn[(size_t)b * kH + hcol] = hb;
      outs[((size_t)s * kB + b) * kH + hcol] = hb;
      if (s == kSEQ - 1) {
        hT[(size_t)b * kH + hcol] = h;
        cT[(size_t)b * kH + hcol] = c;
      }
    }

    // ---- arrive (release our h writes), then hide next step's X-part in the wait shadow ----
    __threadfence();
    __syncthreads();
    if (t == 0) atomicAdd(ctr, 1u);

    aX0 = vzero; aX1 = vzero;
    if (s + 1 < kSEQ) {
      const __bf16* arow = Xb + (size_t)(s + 1) * (kB * kE) + arowoff;
#pragma unroll
      for (int kk = 0; kk < 8; ++kk) {
        v16bf a = load_a16(arow + kk * 32);
        aX0 = wmma_bf16(a, lds_b(Bsw, kk * 4 + nb, lane), aX0);
        aX1 = wmma_bf16(a, lds_b(Bsw, kk * 4 + nb + 1, lane), aX1);
      }
      if (s + 2 < kSEQ) {                          // prefetch step s+2 X rows into cache
        const __bf16* pf = Xb + (size_t)(s + 2) * (kB * kE)
                              + (size_t)(mi * 16 + ml) * kE + (size_t)hi * 128;
        __builtin_prefetch(pf, 0, 3);
        __builtin_prefetch(pf + 64, 0, 3);
      }
    }
  }
}

__global__ void attn_vec_k(const float* __restrict__ attn_w, const __bf16* __restrict__ outs,
                           float* __restrict__ av) {
  int h = threadIdx.x;                                        // 512 threads
  const __bf16* q = outs + ((size_t)(kSEQ - 1) * kB + (kB - 1)) * kH;
  float s = 0.f;
  for (int k = 0; k < kH; ++k) s += attn_w[(size_t)k * kH + h] * (float)q[k];
  av[h] = s;
}

__global__ void attn_scores_k(const __bf16* __restrict__ outs, const float* __restrict__ av,
                              float* __restrict__ sc) {
  int gid = blockIdx.x * 256 + threadIdx.x;
  if (gid >= (kSEQ - 1) * kB) return;
  const __bf16* row = outs + (size_t)gid * kH;
  float acc = 0.f;
  for (int h = 0; h < kH; ++h) acc += (float)row[h] * av[h];
  sc[gid] = acc;
}

__global__ void attn_softmax_k(float* __restrict__ sc) {
  int b = blockIdx.x, t = threadIdx.x;
  __shared__ float red[256];
  float mx = -3.4e38f;
  for (int s = t; s < kSEQ - 1; s += 256) mx = fmaxf(mx, sc[s * kB + b]);
  red[t] = mx; __syncthreads();
  for (int o = 128; o > 0; o >>= 1) { if (t < o) red[t] = fmaxf(red[t], red[t + o]); __syncthreads(); }
  mx = red[0]; __syncthreads();
  float sm = 0.f;
  for (int s = t; s < kSEQ - 1; s += 256) sm += __expf(sc[s * kB + b] - mx);
  red[t] = sm; __syncthreads();
  for (int o = 128; o > 0; o >>= 1) { if (t < o) red[t] += red[t + o]; __syncthreads(); }
  float inv = 1.0f / red[0];
  for (int s = t; s < kSEQ - 1; s += 256) sc[s * kB + b] = __expf(sc[s * kB + b] - mx) * inv;
}

__global__ void attn_ctx_k(const float* __restrict__ sc, const __bf16* __restrict__ outs,
                           __bf16* __restrict__ cat) {
  int gid = blockIdx.x * 256 + threadIdx.x;                   // B*H threads
  int b = gid >> 9, h = gid & 511;
  float acc = 0.f;
  for (int s = 0; s < kSEQ - 1; ++s)
    acc += sc[s * kB + b] * (float)outs[((size_t)s * kB + b) * kH + h];
  cat[(size_t)b * (2 * kH) + h] = (__bf16)acc;
  cat[(size_t)b * (2 * kH) + kH + h] = outs[((size_t)(kSEQ - 1) * kB + b) * kH + h];
}

__global__ void logits_wmma_k(const __bf16* __restrict__ cat, const float* __restrict__ h2o_w,
                              const float* __restrict__ h2o_b, float* __restrict__ out) {
  const int lane = threadIdx.x & 31, wave = threadIdx.x >> 5;
  const int hi = lane >> 4, ml = lane & 15;
  const int mi = wave >> 1;
  const int nt0 = blockIdx.x * 4 + (wave & 1) * 2;            // 625 valid 16-wide N tiles
  const int n0 = nt0 * 16 + ml, n1 = (nt0 + 1) * 16 + ml;
  const bool v0 = (nt0 < 625), v1 = (nt0 + 1 < 625);          // uniform per wave
  v8f acc0 = {}, acc1 = {};
  const __bf16* arow  = cat + (size_t)(mi * 16 + ml) * (2 * kH) + hi * 8;
  const float*  brow0 = h2o_w + (size_t)(v0 ? n0 : 0) * (2 * kH) + hi * 8;
  const float*  brow1 = h2o_w + (size_t)(v1 ? n1 : 0) * (2 * kH) + hi * 8;
#pragma unroll 4
  for (int kk = 0; kk < 32; ++kk) {                           // K = 1024
    v16bf a  = load_a16(arow + kk * 32);
    v16bf b0 = load_b_f32(brow0 + kk * 32);
    v16bf b1 = load_b_f32(brow1 + kk * 32);
    acc0 = wmma_bf16(a, b0, acc0);
    acc1 = wmma_bf16(a, b1, acc1);
  }
#pragma unroll
  for (int r = 0; r < 8; ++r) {
    int row = mi * 16 + r + hi * 8;
    if (v0) out[(size_t)row * kV + n0] = acc0[r] + h2o_b[n0];
    if (v1) out[(size_t)row * kV + n1] = acc1[r] + h2o_b[n1];
  }
}

__global__ void log_softmax_k(float* __restrict__ out) {
  int b = blockIdx.x, t = threadIdx.x;
  float* row = out + (size_t)b * kV;
  __shared__ float red[256];
  float mx = -3.4e38f;
  for (int n = t; n < kV; n += 256) mx = fmaxf(mx, row[n]);
  red[t] = mx; __syncthreads();
  for (int o = 128; o > 0; o >>= 1) { if (t < o) red[t] = fmaxf(red[t], red[t + o]); __syncthreads(); }
  mx = red[0]; __syncthreads();
  float sm = 0.f;
  for (int n = t; n < kV; n += 256) sm += __expf(row[n] - mx);
  red[t] = sm; __syncthreads();
  for (int o = 128; o > 0; o >>= 1) { if (t < o) red[t] += red[t + o]; __syncthreads(); }
  float lse = mx + __logf(red[0]);
  __syncthreads();
  for (int n = t; n < kV; n += 256) row[n] = row[n] - lse;
}

// ---------------- launch ----------------
extern "C" void kernel_launch(void* const* d_in, const int* in_sizes, int n_in,
                              void* d_out, int out_size, void* d_ws, size_t ws_size,
                              hipStream_t stream) {
  (void)in_sizes; (void)n_in; (void)out_size; (void)ws_size;
  const int*   idx    = (const int*)d_in[0];
  const float* h0     = (const float*)d_in[1];
  const float* c0     = (const float*)d_in[2];
  const float* emb    = (const float*)d_in[3];
  const float* w_ih   = (const float*)d_in[4];
  const float* w_hh   = (const float*)d_in[5];
  const float* b_ih   = (const float*)d_in[6];
  const float* b_hh   = (const float*)d_in[7];
  const float* attn_w = (const float*)d_in[8];
  const float* h2o_w  = (const float*)d_in[9];
  const float* h2o_b  = (const float*)d_in[10];

  float* out  = (float*)d_out;
  float* logp = out;
  float* hT   = out + (size_t)kB * kV;
  float* cT   = hT + (size_t)kB * kH;

  char* ws = (char*)d_ws;
  unsigned* ctr = (unsigned*)(ws + CTR_OFF);
  __bf16* Xb   = (__bf16*)(ws + XB_OFF);
  __bf16* outs = (__bf16*)(ws + OUTS_OFF);
  __bf16* Hb   = (__bf16*)(ws + HB_OFF);
  float*  av   = (float*)(ws + AV_OFF);
  float*  sc   = (float*)(ws + SC_OFF);
  __bf16* cat  = (__bf16*)(ws + CAT_OFF);

  hipMemsetAsync(ctr, 0, 256, stream);
  embed_bf16_k<<<(kSEQ * kB * (kE / 4)) / 256, 256, 0, stream>>>(idx, emb, Xb);
  init_h_k<<<(kB * kH) / 256, 256, 0, stream>>>(h0, Hb);

  (void)hipFuncSetAttribute(reinterpret_cast<const void*>(lstm_persistent_k),
                            hipFuncAttributeMaxDynamicSharedMemorySize, 114688);
  lstm_persistent_k<<<kNWG, 256, 114688, stream>>>(w_ih, w_hh, b_ih, b_hh, c0,
                                                   Xb, Hb, outs, hT, cT, ctr);

  attn_vec_k<<<1, 512, 0, stream>>>(attn_w, outs, av);
  attn_scores_k<<<256, 256, 0, stream>>>(outs, av, sc);
  attn_softmax_k<<<kB, 256, 0, stream>>>(sc);
  attn_ctx_k<<<(kB * kH) / 256, 256, 0, stream>>>(sc, outs, cat);
  logits_wmma_k<<<157, 256, 0, stream>>>(cat, h2o_w, h2o_b, logp);
  log_softmax_k<<<kB, 256, 0, stream>>>(logp);
}
